// MiniMeshGraphNet_10393820856379
// MI455X (gfx1250) — matrix operand order repaired and checked by reading, not compiled, not run on hardware
//
#include <hip/hip_runtime.h>
#include <hip/hip_bf16.h>

// ---------------------------------------------------------------------------
// MiniMeshGraphNet forward for gfx1250 (MI455X).
// bf16 WMMA (v_wmma_f32_16x16x32_bf16) for every GEMM; weights pre-swizzled
// into B-fragment order (one contiguous 32B load per lane); activations
// staged fp32->bf16 into LDS with bank-conflict-free padded strides so A
// fragments are two conflict-free ds_load_b128 per lane; kt-outer/nt-inner
// WMMA ordering gives 4 independent WMMAs per B-load clause; segment_sum via
// coalesced global_atomic_add_f32.
// ---------------------------------------------------------------------------

typedef __bf16 bf16_t;
typedef bf16_t v16bf __attribute__((ext_vector_type(16)));
typedef bf16_t v8bf  __attribute__((ext_vector_type(8)));
typedef float  v8f   __attribute__((ext_vector_type(8)));

#define WAVES 4
#define LN_EPS 1e-5f
#define SH_STRIDE 72   // bf16 elems/row for the hidden tile (36 dwords: conflict-free)
#define SY_STRIDE 68   // f32  elems/row for the y tile     (68 dwords: conflict-free LN)

__device__ __forceinline__ v8f wmma_bf16(v16bf a, v16bf b, v8f c) {
  // 8 args: (neg_a, A, neg_b, B, c_mod, C, reuse_a, reuse_b)
  return __builtin_amdgcn_wmma_f32_16x16x32_bf16(false, a, false, b, (short)0, c,
                                                 false, false);
}

// A fragment (16x32 bf16, M x K) from a row-major LDS tile.
// Lane layout: lanes 0-15: M=lane, K = 0..15;
//              lanes 16-31: M=lane-16, K = 16..31.
// stride must be a multiple of 8 bf16 (16B alignment for v8bf loads).
__device__ __forceinline__ v16bf load_a_frag(const bf16_t* tile, int lane, int kt,
                                             int stride) {
  const int m  = lane & 15;
  const int k0 = kt * 32 + ((lane & 16) ? 16 : 0);
  const bf16_t* p = tile + m * stride + k0;
  union { v16bf v; v8bf h[2]; } u;
  u.h[0] = *(const v8bf*)(p);
  u.h[1] = *(const v8bf*)(p + 8);
  return u.v;
}

// B fragment from pre-swizzled weights: frag(kt,nt) = 32 lanes x 16 bf16 contiguous.
__device__ __forceinline__ v16bf load_b_frag(const bf16_t* wswz, int kt, int nt,
                                             int ntiles, int lane) {
  return ((const v16bf*)wswz)[((kt * ntiles) + nt) * 32 + lane];
}

__device__ __forceinline__ float silu_f(float x) {
  return x * (1.0f / (1.0f + __expf(-x)));
}

// ---------------------------------------------------------------------------
// Weight pre-swizzle: row-major fp32 W[Kreal][Nreal] -> bf16 fragment order,
// zero-padded to Ktiles*32 x Ntiles*16.
// B layout (this code's contiguous k order, matching load_a_frag): lane holds col N = nt*16 + lane%16,
// rows K = kt*32 + (lane>=16 ? 16 : 0) + i, i = 0..15.
// ---------------------------------------------------------------------------
__global__ void swizzle_w_kernel(const float* __restrict__ w, bf16_t* __restrict__ dst,
                                 int Kreal, int Nreal, int Ktiles, int Ntiles) {
  const int idx = blockIdx.x * blockDim.x + threadIdx.x;
  const int total = Ktiles * Ntiles * 512;
  if (idx >= total) return;
  const int kt   = idx / (Ntiles * 512);
  const int rem  = idx % (Ntiles * 512);
  const int nt   = rem / 512;
  const int e    = rem % 512;
  const int lane = e >> 4;
  const int i    = e & 15;
  const int n = nt * 16 + (lane & 15);
  const int k = kt * 32 + ((lane & 16) ? 16 : 0) + i;
  float v = (k < Kreal && n < Nreal) ? w[k * Nreal + n] : 0.0f;
  dst[idx] = (bf16_t)v;
}

__global__ void fill0_kernel(float* __restrict__ p, int n) {
  const int i = blockIdx.x * blockDim.x + threadIdx.x;
  if (i < n) p[i] = 0.0f;
}

// ---------------------------------------------------------------------------
// Fused MLP block: y = LN( W2*silu(W1*in + b1) + b2 + Wres*in + bres )
// MODE 0: in = src0 rows [M x in_dim], zero-padded to KT*32 (encoders)
// MODE 1: in = concat(src0[row], src1[row])           (node MLP, 128)
// MODE 2: in = concat(src0[row], src1[srcI], src1[dstI]) (edge MLP, 192)
//         + scatter-add of normalized output into agg[dstI]
// One wave per 16-row tile; 4 waves per block.
// ---------------------------------------------------------------------------
struct MlpArgs {
  const float* src0;
  const float* src1;
  const int*   src_idx;
  const int*   dst_idx;
  int in_dim;
  int M;
  const bf16_t* w1s; const bf16_t* w2s; const bf16_t* wrs;
  const float* b1; const float* b2; const float* br;
  const float* gamma; const float* beta;
  float* out;
  float* agg;
};

template<int MODE, int KT>
__global__ __launch_bounds__(WAVES * 32) void mlp_block_kernel(MlpArgs a) {
  constexpr int KPAD = KT * 32;
  constexpr int SIN  = KPAD + 8;  // padded stride: (KPAD+8)/2 dwords/row, conflict-free
  __shared__ bf16_t s_in[WAVES][16 * SIN];
  __shared__ bf16_t s_h [WAVES][16 * SH_STRIDE];
  __shared__ float  s_y [WAVES][16 * SY_STRIDE];

  const int w     = threadIdx.x >> 5;
  const int lane  = threadIdx.x & 31;
  const int tile0 = (blockIdx.x * WAVES + w) * 16;
  if (tile0 >= a.M) return;  // wave-uniform early out (EXEC stays all-ones)

  // ---- stage input tile (fp32 -> bf16) row-major [16][SIN] in LDS
  for (int r = 0; r < 16; ++r) {
    const int row = tile0 + r;
    const bool ok = row < a.M;
    if (MODE == 0) {
      for (int c = lane; c < KPAD; c += 32) {
        float v = (ok && c < a.in_dim) ? a.src0[(size_t)row * a.in_dim + c] : 0.0f;
        s_in[w][r * SIN + c] = (bf16_t)v;
      }
    } else if (MODE == 1) {
      for (int c = lane; c < KPAD; c += 32) {
        float v = 0.0f;
        if (ok) v = (c < 64) ? a.src0[(size_t)row * 64 + c]
                             : a.src1[(size_t)row * 64 + (c - 64)];
        s_in[w][r * SIN + c] = (bf16_t)v;
      }
    } else {
      const int si = ok ? a.src_idx[row] : 0;
      const int di = ok ? a.dst_idx[row] : 0;
      for (int c = lane; c < KPAD; c += 32) {
        float v = 0.0f;
        if (ok) {
          if (c < 64)       v = a.src0[(size_t)row * 64 + c];
          else if (c < 128) v = a.src1[(size_t)si * 64 + (c - 64)];
          else              v = a.src1[(size_t)di * 64 + (c - 128)];
        }
        s_in[w][r * SIN + c] = (bf16_t)v;
      }
    }
  }

  const int n16 = lane & 15;
  const int mb  = (lane & 16) ? 8 : 0;  // C/D layout: lanes 16-31 hold M=8..15

  // ---- h = silu(in @ W1 + b1) -> bf16 [16][SH_STRIDE] in LDS
  {
    v8f acc[4];
#pragma unroll
    for (int nt = 0; nt < 4; ++nt) {
      const float bv = a.b1[nt * 16 + n16];
#pragma unroll
      for (int i = 0; i < 8; ++i) acc[nt][i] = bv;
    }
#pragma unroll
    for (int kt = 0; kt < KT; ++kt) {
      const v16bf af = load_a_frag(s_in[w], lane, kt, SIN);
#pragma unroll
      for (int nt = 0; nt < 4; ++nt)
        acc[nt] = wmma_bf16(af, load_b_frag(a.w1s, kt, nt, 4, lane), acc[nt]);
    }
#pragma unroll
    for (int nt = 0; nt < 4; ++nt)
#pragma unroll
      for (int v = 0; v < 8; ++v)
        s_h[w][(mb + v) * SH_STRIDE + nt * 16 + n16] = (bf16_t)silu_f(acc[nt][v]);
  }

  // ---- y = h @ W2 + b2 + in @ Wres + bres -> f32 [16][SY_STRIDE] in LDS
  {
    v8f acc[4];
#pragma unroll
    for (int nt = 0; nt < 4; ++nt) {
      const float bv = a.b2[nt * 16 + n16] + a.br[nt * 16 + n16];
#pragma unroll
      for (int i = 0; i < 8; ++i) acc[nt][i] = bv;
    }
#pragma unroll
    for (int kt = 0; kt < 2; ++kt) {
      const v16bf ah = load_a_frag(s_h[w], lane, kt, SH_STRIDE);
#pragma unroll
      for (int nt = 0; nt < 4; ++nt)
        acc[nt] = wmma_bf16(ah, load_b_frag(a.w2s, kt, nt, 4, lane), acc[nt]);
    }
#pragma unroll
    for (int kt = 0; kt < KT; ++kt) {
      const v16bf ar = load_a_frag(s_in[w], lane, kt, SIN);
#pragma unroll
      for (int nt = 0; nt < 4; ++nt)
        acc[nt] = wmma_bf16(ar, load_b_frag(a.wrs, kt, nt, 4, lane), acc[nt]);
    }
#pragma unroll
    for (int nt = 0; nt < 4; ++nt)
#pragma unroll
      for (int v = 0; v < 8; ++v)
        s_y[w][(mb + v) * SY_STRIDE + nt * 16 + n16] = acc[nt][v];
  }

  // ---- LayerNorm over 64 features (one row per lane 0..15), write out,
  //      keep normalized values in LDS for the scatter below.
  if (lane < 16) {
    const int r = lane;
    const int row = tile0 + r;
    float sum = 0.0f, sq = 0.0f;
    for (int c = 0; c < 64; ++c) {
      const float v = s_y[w][r * SY_STRIDE + c];
      sum += v; sq += v * v;
    }
    const float mu  = sum * (1.0f / 64.0f);
    const float var = sq * (1.0f / 64.0f) - mu * mu;
    const float rs  = rsqrtf(var + LN_EPS);
    for (int c = 0; c < 64; ++c) {
      const float v = (s_y[w][r * SY_STRIDE + c] - mu) * rs * a.gamma[c] + a.beta[c];
      s_y[w][r * SY_STRIDE + c] = v;
      if (row < a.M) a.out[(size_t)row * 64 + c] = v;
    }
  }

  // ---- segment_sum scatter (edge layers): agg[dst] += e_new, coalesced atomics
  if (MODE == 2) {
    for (int r = 0; r < 16; ++r) {
      const int row = tile0 + r;
      if (row < a.M) {
        const int d = a.dst_idx[row];
        atomicAdd(&a.agg[(size_t)d * 64 + lane],      s_y[w][r * SY_STRIDE + lane]);
        atomicAdd(&a.agg[(size_t)d * 64 + lane + 32], s_y[w][r * SY_STRIDE + lane + 32]);
      }
    }
  }
}

// ---------------------------------------------------------------------------
// Decoder: out = silu(x @ W1 + b1) @ W2 + b2   (64 -> 64 -> 2, W2 padded to 16 cols)
// ---------------------------------------------------------------------------
struct DecArgs {
  const float* x; int M;
  const bf16_t* w1s; const float* b1;
  const bf16_t* w2s; const float* b2;
  float* out;
};

__global__ __launch_bounds__(WAVES * 32) void decoder_kernel(DecArgs a) {
  __shared__ bf16_t s_in[WAVES][16 * SH_STRIDE];
  __shared__ bf16_t s_h [WAVES][16 * SH_STRIDE];
  const int w     = threadIdx.x >> 5;
  const int lane  = threadIdx.x & 31;
  const int tile0 = (blockIdx.x * WAVES + w) * 16;
  if (tile0 >= a.M) return;

  for (int r = 0; r < 16; ++r) {
    const int row = tile0 + r;
    for (int c = lane; c < 64; c += 32) {
      float v = (row < a.M) ? a.x[(size_t)row * 64 + c] : 0.0f;
      s_in[w][r * SH_STRIDE + c] = (bf16_t)v;
    }
  }
  const int n16 = lane & 15;
  const int mb  = (lane & 16) ? 8 : 0;
  {
    v8f acc[4];
#pragma unroll
    for (int nt = 0; nt < 4; ++nt) {
      const float bv = a.b1[nt * 16 + n16];
#pragma unroll
      for (int i = 0; i < 8; ++i) acc[nt][i] = bv;
    }
#pragma unroll
    for (int kt = 0; kt < 2; ++kt) {
      const v16bf af = load_a_frag(s_in[w], lane, kt, SH_STRIDE);
#pragma unroll
      for (int nt = 0; nt < 4; ++nt)
        acc[nt] = wmma_bf16(af, load_b_frag(a.w1s, kt, nt, 4, lane), acc[nt]);
    }
#pragma unroll
    for (int nt = 0; nt < 4; ++nt)
#pragma unroll
      for (int v = 0; v < 8; ++v)
        s_h[w][(mb + v) * SH_STRIDE + nt * 16 + n16] = (bf16_t)silu_f(acc[nt][v]);
  }
  {
    const float bv = (n16 < 2) ? a.b2[n16] : 0.0f;
    v8f acc;
#pragma unroll
    for (int i = 0; i < 8; ++i) acc[i] = bv;
#pragma unroll
    for (int kt = 0; kt < 2; ++kt)
      acc = wmma_bf16(load_a_frag(s_h[w], lane, kt, SH_STRIDE),
                      load_b_frag(a.w2s, kt, 0, 1, lane), acc);
    if (n16 < 2) {
#pragma unroll
      for (int v = 0; v < 8; ++v) {
        const int row = tile0 + mb + v;
        if (row < a.M) a.out[(size_t)row * 2 + n16] = acc[v];
      }
    }
  }
}

// ---------------------------------------------------------------------------
// Host orchestration.
// Input order: node_features, edge_features, edge_index, then the params
// pytree flattened JAX-style (dict keys sorted):
//   blocks[0..3]: edge_mlp{beta,gamma,l1.b,l1.w,l2.b,l2.w,res.b,res.w},
//                 node_mlp{same 8}        -> 16 leaves per block
//   dec1{b,w}, dec2{b,w}, edge_enc{8 leaves}, node_enc{8 leaves}
// ---------------------------------------------------------------------------
extern "C" void kernel_launch(void* const* d_in, const int* in_sizes, int n_in,
                              void* d_out, int out_size, void* d_ws, size_t ws_size,
                              hipStream_t stream) {
  (void)in_sizes; (void)n_in; (void)out_size; (void)ws_size;
  const int N = 50000, E = 400000, H = 64;
  const float* node_f = (const float*)d_in[0];
  const float* edge_f = (const float*)d_in[1];
  const int*   eidx   = (const int*)d_in[2];
  const int* srcI = eidx;
  const int* dstI = eidx + E;

  auto P = [&](int i) { return (const float*)d_in[i]; };
  const int PB = 3;
  const int DEC1 = PB + 64, DEC2 = PB + 66, EENC = PB + 68, NENC = PB + 76;

  char* ws = (char*)d_ws;
  size_t off = 0;
  auto carve = [&](size_t bytes) -> void* {
    void* p = ws + off;
    off += (bytes + 255) & ~(size_t)255;
    return p;
  };
  float* x0  = (float*)carve((size_t)N * H * 4);
  float* x1  = (float*)carve((size_t)N * H * 4);
  float* e0  = (float*)carve((size_t)E * H * 4);
  float* e1  = (float*)carve((size_t)E * H * 4);
  float* agg = (float*)carve((size_t)N * H * 4);

  auto swz = [&](const float* wsrc, int Kr, int Nr, int Kt, int Nt) -> bf16_t* {
    const int total = Kt * Nt * 512;
    bf16_t* d = (bf16_t*)carve((size_t)total * 2);
    swizzle_w_kernel<<<(total + 255) / 256, 256, 0, stream>>>(wsrc, d, Kr, Nr, Kt, Nt);
    return d;
  };

  // one-time (per call) weight prep into fragment order
  bf16_t* ne_w1 = swz(P(NENC + 3),   5, 64, 1, 4);
  bf16_t* ne_w2 = swz(P(NENC + 5),  64, 64, 2, 4);
  bf16_t* ne_wr = swz(P(NENC + 7),   5, 64, 1, 4);
  bf16_t* ee_w1 = swz(P(EENC + 3),   2, 64, 1, 4);
  bf16_t* ee_w2 = swz(P(EENC + 5),  64, 64, 2, 4);
  bf16_t* ee_wr = swz(P(EENC + 7),   2, 64, 1, 4);
  bf16_t* bw[4][6];
  for (int l = 0; l < 4; ++l) {
    const int EM = PB + 16 * l, NM = EM + 8;
    bw[l][0] = swz(P(EM + 3), 192, 64, 6, 4);
    bw[l][1] = swz(P(EM + 5),  64, 64, 2, 4);
    bw[l][2] = swz(P(EM + 7), 192, 64, 6, 4);
    bw[l][3] = swz(P(NM + 3), 128, 64, 4, 4);
    bw[l][4] = swz(P(NM + 5),  64, 64, 2, 4);
    bw[l][5] = swz(P(NM + 7), 128, 64, 4, 4);
  }
  bf16_t* d1w = swz(P(DEC1 + 1), 64, 64, 2, 4);
  bf16_t* d2w = swz(P(DEC2 + 1), 64,  2, 2, 1);

  // encoders
  {
    MlpArgs na{};
    na.src0 = node_f; na.in_dim = 5; na.M = N;
    na.w1s = ne_w1; na.w2s = ne_w2; na.wrs = ne_wr;
    na.b1 = P(NENC + 2); na.b2 = P(NENC + 4); na.br = P(NENC + 6);
    na.gamma = P(NENC + 1); na.beta = P(NENC + 0);
    na.out = x0;
    mlp_block_kernel<0, 1><<<(N + 63) / 64, WAVES * 32, 0, stream>>>(na);
  }
  {
    MlpArgs ea{};
    ea.src0 = edge_f; ea.in_dim = 2; ea.M = E;
    ea.w1s = ee_w1; ea.w2s = ee_w2; ea.wrs = ee_wr;
    ea.b1 = P(EENC + 2); ea.b2 = P(EENC + 4); ea.br = P(EENC + 6);
    ea.gamma = P(EENC + 1); ea.beta = P(EENC + 0);
    ea.out = e0;
    mlp_block_kernel<0, 1><<<(E + 63) / 64, WAVES * 32, 0, stream>>>(ea);
  }

  // message-passing layers
  float *x = x0, *xn = x1, *eb = e0, *en = e1;
  for (int l = 0; l < 4; ++l) {
    const int EM = PB + 16 * l, NM = EM + 8;
    fill0_kernel<<<(N * H + 255) / 256, 256, 0, stream>>>(agg, N * H);
    {
      MlpArgs ea{};
      ea.src0 = eb; ea.src1 = x; ea.src_idx = srcI; ea.dst_idx = dstI; ea.M = E;
      ea.w1s = bw[l][0]; ea.w2s = bw[l][1]; ea.wrs = bw[l][2];
      ea.b1 = P(EM + 2); ea.b2 = P(EM + 4); ea.br = P(EM + 6);
      ea.gamma = P(EM + 1); ea.beta = P(EM + 0);
      ea.out = en; ea.agg = agg;
      mlp_block_kernel<2, 6><<<(E + 63) / 64, WAVES * 32, 0, stream>>>(ea);
    }
    {
      MlpArgs na{};
      na.src0 = x; na.src1 = agg; na.M = N;
      na.w1s = bw[l][3]; na.w2s = bw[l][4]; na.wrs = bw[l][5];
      na.b1 = P(NM + 2); na.b2 = P(NM + 4); na.br = P(NM + 6);
      na.gamma = P(NM + 1); na.beta = P(NM + 0);
      na.out = xn;
      mlp_block_kernel<1, 4><<<(N + 63) / 64, WAVES * 32, 0, stream>>>(na);
    }
    float* t;
    t = eb; eb = en; en = t;
    t = x;  x  = xn; xn = t;
  }

  // decoder
  {
    DecArgs da{};
    da.x = x; da.M = N;
    da.w1s = d1w; da.b1 = P(DEC1 + 0);
    da.w2s = d2w; da.b2 = P(DEC2 + 0);
    da.out = (float*)d_out;
    decoder_kernel<<<(N + 63) / 64, WAVES * 32, 0, stream>>>(da);
  }
}
